// PatchFlow_43937515438246
// MI455X (gfx1250) — matrix operand order repaired and checked by reading, not compiled, not run on hardware
//
#include <hip/hip_runtime.h>
#include <hip/hip_bf16.h>
#include <math.h>

#define DD 8
#define HID 128
#define CTX 64
#define NF 4
#define NPOS 1024
#define NROW 32768

typedef __attribute__((ext_vector_type(16))) _Float16 v16h;
typedef __attribute__((ext_vector_type(8)))  float    v8f;

union F16x16 { v16h v; float4 q[2]; _Float16 h[16]; };

// per-flow packed f16 weight layout (halves)
// W0p is stored in full per-lane fragment form: [j(8)][lane(32)][16 halves]
#define W0P_OFF   0
#define W1_OFF    4096
#define W2_OFF    36864
#define WOUT_OFF  69632
#define FLOWSZ_H  71680   // 143360 bytes

// dynamic LDS layout (bytes)
#define LDS_ACT_OFF   143360            // 4 waves * 2048 halves
#define LDS_Z_OFF     159744            // 64*8 f32
#define LDS_XIN_OFF   161792            // 64*8 f32
#define LDS_O_OFF     163840            // 4*256 f32
#define LDS_ZERO_OFF  167936            // 64 B of zeros
#define LDS_TOTAL     168000

static __device__ __forceinline__ float sigmoidf_(float x) { return 1.f / (1.f + expf(-x)); }

static __device__ __forceinline__ v16h ldsA(const _Float16* rowp, int kb, int hi) {
  F16x16 u;
  u.q[0] = *(const float4*)(rowp + kb + hi * 8);
  u.q[1] = *(const float4*)(rowp + kb + 16 + hi * 8);
  return u.v;
}
static __device__ __forceinline__ v16h ldsB(const _Float16* w, int nrow, int kb, int hi) {
  const _Float16* p = w + nrow * 128 + kb + hi * 8;
  F16x16 u;
  u.q[0] = *(const float4*)(p);
  u.q[1] = *(const float4*)(p + 16);
  return u.v;
}
static __device__ __forceinline__ v16h ldsFrag(const _Float16* p) {   // 16 contiguous halves
  F16x16 u;
  u.q[0] = *(const float4*)(p);
  u.q[1] = *(const float4*)(p + 8);
  return u.v;
}
static __device__ __forceinline__ v8f wmma16(v16h a, v16h b, v8f c) {
  return __builtin_amdgcn_wmma_f32_16x16x32_f16(false, a, false, b, (short)0, c, false, false);
}
static __device__ __forceinline__ void lds_fence() { asm volatile("" ::: "memory"); }

// stage acc + per-tile bias (addv[j]) into the f16 activation tile
static __device__ __forceinline__ void stage_act(_Float16* actW, const v8f* acc,
                                                 const float* addv,
                                                 int n, int hi, bool relu) {
#pragma unroll
  for (int j = 0; j < 8; ++j) {
#pragma unroll
    for (int r = 0; r < 8; ++r) {
      float vv = acc[j][r] + addv[j];
      if (relu) vv = vv > 0.f ? vv : 0.f;
      actW[(r + 8 * hi) * 128 + j * 16 + n] = (_Float16)vv;
    }
  }
  lds_fence();   // keep LDS stores ordered before the fragment loads (wave-local)
}

// 128x128 masked GEMM: first K-chunk writes with C = 0 (no init splats)
static __device__ __forceinline__ void gemm128z(v8f* acc, const _Float16* actRow,
                                                const _Float16* wlds, int n, int hi) {
  const v8f zc = {};
  {
    v16h a = ldsA(actRow, 0, hi);
#pragma unroll
    for (int j = 0; j < 8; ++j) {
      v16h b = ldsB(wlds, j * 16 + n, 0, hi);
      acc[j] = wmma16(a, b, zc);
    }
  }
#pragma unroll
  for (int kc = 1; kc < 4; ++kc) {
    v16h a = ldsA(actRow, kc * 32, hi);
#pragma unroll
    for (int j = 0; j < 8; ++j) {
      v16h b = ldsB(wlds, j * 16 + n, kc * 32, hi);
      acc[j] = wmma16(a, b, acc[j]);
    }
  }
}

// ---------------------------------------------------------------------------
// Kernel 1: depthwise 3x3 stride-2 energy conv -> pattern rows; zero ldj
// ---------------------------------------------------------------------------
__global__ void patch_kernel(const float* __restrict__ x, float* __restrict__ z0,
                             float* __restrict__ ldj0) {
  int gid = blockIdx.x * 256 + threadIdx.x;          // 262144 = NROW * 8
  int row = gid >> 3, c = gid & 7;
  int b = row & 31, pos = row >> 5;
  int oi = pos >> 5, oj = pos & 31;
  float acc = 0.f;
#pragma unroll
  for (int di = 0; di < 3; ++di) {
#pragma unroll
    for (int dj = 0; dj < 3; ++dj) {
      int y = 2 * oi + di - 1, xc = 2 * oj + dj - 1;
      if (y >= 0 && y < 64 && xc >= 0 && xc < 64) {
        float v = x[((b * 8 + c) * 64 + y) * 64 + xc];
        acc += v * v;
      }
    }
  }
  z0[row * 8 + c] = sqrtf(acc);
  if (c == 0) ldj0[row] = 0.f;
}

// ---------------------------------------------------------------------------
// Kernel 2: masked f16 weight pack + LU matrices  (one block per flow)
// ---------------------------------------------------------------------------
__global__ void prep_kernel(const float* __restrict__ W0, const float* __restrict__ Wl1,
                            const float* __restrict__ Wl2, const float* __restrict__ Wout,
                            const float* __restrict__ lu_lower, const float* __restrict__ lu_upper,
                            const float* __restrict__ lu_udiag, const float* __restrict__ lu_bias,
                            _Float16* __restrict__ wpack, float* __restrict__ lupack) {
  int f = blockIdx.x, tid = threadIdx.x;
  _Float16* wp = wpack + (size_t)f * FLOWSZ_H;
  for (int i = tid; i < 4096; i += 256) {            // W0 * M_IN, per-lane fragment form
    int j = i >> 9, ln = (i >> 4) & 31, hh = i & 15;
    int n = ln & 15, hi = ln >> 4;
    float v = 0.f;
    if (hi == 0 && hh < 8) {
      int ng = j * 16 + n;
      int hd = ng % 7 + 1;
      if (hd >= hh + 1) v = W0[(f * 128 + ng) * 8 + hh];
    }
    wp[W0P_OFF + i] = (_Float16)v;
  }
  for (int blk = 0; blk < 2; ++blk) {
    for (int i = tid; i < 16384; i += 256) {         // Wl* * M_H, [ng][k]
      int ng = i >> 7, k = i & 127;
      float m = (ng % 7 + 1 >= k % 7 + 1) ? 1.f : 0.f;
      wp[W1_OFF + blk * 16384 + i] = (_Float16)(m * Wl1[(((size_t)f * 2 + blk) * 128 + ng) * 128 + k]);
      wp[W2_OFF + blk * 16384 + i] = (_Float16)(m * Wl2[(((size_t)f * 2 + blk) * 128 + ng) * 128 + k]);
    }
  }
  for (int i = tid; i < 2048; i += 256) {            // Wout * M_OUT, [ng(16)][k]
    int ng = i >> 7, k = i & 127;
    float m = (ng / 2 + 1 > k % 7 + 1) ? 1.f : 0.f;
    wp[WOUT_OFF + i] = (_Float16)(m * Wout[((size_t)f * 16 + ng) * 128 + k]);
  }
  if (tid == 0) {                                    // A = L@U, bias, logdet(U)
    float L[8][8], U[8][8];
    for (int i = 0; i < 8; ++i)
      for (int j = 0; j < 8; ++j) { L[i][j] = (i == j) ? 1.f : 0.f; U[i][j] = 0.f; }
    for (int i = 1; i < 8; ++i)
      for (int j = 0; j < i; ++j) L[i][j] = lu_lower[f * 28 + i * (i - 1) / 2 + j];
    float logdet = 0.f;
    for (int i = 0; i < 8; ++i) {
      float ud = log1pf(expf(lu_udiag[f * 8 + i])) + 0.001f;
      U[i][i] = ud;
      logdet += logf(ud);
      for (int j = i + 1; j < 8; ++j)
        U[i][j] = lu_upper[f * 28 + 7 * i - i * (i - 1) / 2 + (j - i - 1)];
    }
    float* lp = lupack + f * 80;
    for (int i = 0; i < 8; ++i)
      for (int j = 0; j < 8; ++j) {
        float a = 0.f;
        for (int k = 0; k < 8; ++k) a += L[i][k] * U[k][j];
        lp[i * 8 + j] = a;
      }
    for (int i = 0; i < 8; ++i) lp[64 + i] = lu_bias[f * 8 + i];
    lp[72] = logdet;
  }
}

// ---------------------------------------------------------------------------
// Kernel 3: context projections (only 1024 unique ctx rows!)
// ---------------------------------------------------------------------------
__global__ void ctx_kernel(const float* __restrict__ b0, const float* __restrict__ Wc0,
                           const float* __restrict__ bc0, const float* __restrict__ Wcb,
                           const float* __restrict__ bcb, float* __restrict__ c0,
                           float* __restrict__ gsig) {
  int pos = blockIdx.x, f = blockIdx.y, n = threadIdx.x;
  __shared__ float pe[64];
  if (n < 64) {
    int base = (n < 32) ? (pos >> 5) : (pos & 31);
    int cc = n & 31, t = cc >> 1;
    float invf = expf(-((float)(2 * t) / 32.f) * logf(10000.f));
    float s = (float)base * invf;
    pe[n] = (cc & 1) ? cosf(s) : sinf(s);
  }
  __syncthreads();
  float a0 = b0[f * 128 + n] + bc0[f * 128 + n];
  for (int c = 0; c < 64; ++c) a0 += pe[c] * Wc0[((size_t)f * 128 + n) * 64 + c];
  c0[((size_t)f * NPOS + pos) * 128 + n] = a0;
  for (int blk = 0; blk < 2; ++blk) {
    float g = bcb[((size_t)f * 2 + blk) * 128 + n];
    for (int c = 0; c < 64; ++c) g += pe[c] * Wcb[(((size_t)f * 2 + blk) * 128 + n) * 64 + c];
    gsig[(((size_t)f * 2 + blk) * NPOS + pos) * 128 + n] = sigmoidf_(g);
  }
}

// ---------------------------------------------------------------------------
// Kernel 4: one full flow (perm + LU + 8-step AR inversion + ldj), WMMA core.
// Accumulation chains start from C = 0; biases folded into staging/gating.
// ---------------------------------------------------------------------------
__global__ void __launch_bounds__(128, 4)
flow_kernel(float* __restrict__ z_io,
            const float* __restrict__ ldj_in, float* __restrict__ ldj_out,
            float* __restrict__ zfin,
            const _Float16* __restrict__ wbuf_g,
            const float* __restrict__ c0f, const float* __restrict__ gsigf,
            const float* __restrict__ lu_g,
            const float* __restrict__ bl1_g, const float* __restrict__ bl2_g,
            const float* __restrict__ bout_g, const int* __restrict__ perm_g) {
  extern __shared__ char dynsm[];
  _Float16* sW    = (_Float16*)dynsm;
  _Float16* sW0p  = sW + W0P_OFF;
  _Float16* sW1   = sW + W1_OFF;
  _Float16* sW2   = sW + W2_OFF;
  _Float16* sWout = sW + WOUT_OFF;
  _Float16* sAct  = (_Float16*)(dynsm + LDS_ACT_OFF);
  float*    sZ    = (float*)(dynsm + LDS_Z_OFF);
  float*    sXin  = (float*)(dynsm + LDS_XIN_OFF);
  float*    sO    = (float*)(dynsm + LDS_O_OFF);
  float*    sZero = (float*)(dynsm + LDS_ZERO_OFF);

  __shared__ float sBl1[2 * 128], sBl2[2 * 128], sBout[16], sAlu[64], sLuBias[8], sLogdet;
  __shared__ int sPerm[8];

  const int tid = threadIdx.x;
  __builtin_prefetch(wbuf_g, 0, 1);

  // ---- cooperative weight/bias load into LDS ----
  {
    const uint4* src = (const uint4*)wbuf_g;
    uint4* dst = (uint4*)sW;
    for (int i = tid; i < (FLOWSZ_H * 2) / 16; i += 128) dst[i] = src[i];
    for (int i = tid; i < 256; i += 128) { sBl1[i] = bl1_g[i]; sBl2[i] = bl2_g[i]; }
    if (tid < 16) { sBout[tid] = bout_g[tid]; sZero[tid] = 0.f; }
    if (tid < 64) sAlu[tid] = lu_g[tid];
    if (tid < 8) { sLuBias[tid] = lu_g[64 + tid]; sPerm[tid] = perm_g[tid]; }
    if (tid == 0) sLogdet = lu_g[72];
  }
  __syncthreads();

  const int rowbase = blockIdx.x * 64;

  // ---- permute + LU -> xin; z carry = 0 ----
  if (tid < 64) {
    int row = rowbase + tid;
    float zp[8];
#pragma unroll
    for (int j = 0; j < 8; ++j) zp[j] = z_io[(size_t)row * 8 + sPerm[j]];
#pragma unroll
    for (int i = 0; i < 8; ++i) {
      float a = sLuBias[i];
#pragma unroll
      for (int j = 0; j < 8; ++j) a += sAlu[i * 8 + j] * zp[j];
      sXin[tid * 8 + i] = a;
      sZ[tid * 8 + i] = 0.f;
    }
  }
  __syncthreads();

  const int wave = tid >> 5, lane = tid & 31;
  const int n = lane & 15, hi = lane >> 4, m = lane & 15;
  const int pos = (rowbase + wave * 16) >> 5;          // constant per wave

  // per-lane constants -- invariant across all 9 MADE calls
  float cin[8], gsv[2][8], b1v[2][8], b2v[2][8];
#pragma unroll
  for (int j = 0; j < 8; ++j) cin[j] = c0f[(size_t)pos * 128 + j * 16 + n];
#pragma unroll
  for (int blk = 0; blk < 2; ++blk)
#pragma unroll
    for (int j = 0; j < 8; ++j) {
      gsv[blk][j] = gsigf[((size_t)blk * NPOS + pos) * 128 + j * 16 + n];
      b1v[blk][j] = sBl1[blk * 128 + j * 16 + n];
      b2v[blk][j] = sBl2[blk * 128 + j * 16 + n];
    }
  float bov = sBout[n];

  _Float16* actW = sAct + wave * 2048;
  const _Float16* actRow = actW + m * 128;
  float* oW = sO + wave * 256;

  // branchless fragment sources
  const float*    zsrc   = (hi == 0) ? (sZ + (wave * 16 + m) * 8) : sZero;
  const _Float16* b0base = sW0p + lane * 16;           // padded fragments, zeros baked in
  const float4    zero4  = make_float4(0.f, 0.f, 0.f, 0.f);
  const v8f       zc     = {};

  for (int step = 0; step <= DD; ++step) {
    // ---- MADE forward on carry z ----
    // A0 fragment: z padded K 8->32 (zeros for hi==1 / K>=8)
    F16x16 a0;
    {
      float4 zq0 = *(const float4*)(zsrc);
      float4 zq1 = *(const float4*)(zsrc + 4);
      a0.h[0] = (_Float16)zq0.x; a0.h[1] = (_Float16)zq0.y;
      a0.h[2] = (_Float16)zq0.z; a0.h[3] = (_Float16)zq0.w;
      a0.h[4] = (_Float16)zq1.x; a0.h[5] = (_Float16)zq1.y;
      a0.h[6] = (_Float16)zq1.z; a0.h[7] = (_Float16)zq1.w;
      a0.q[1] = zero4;
    }

    v8f h[8];                                        // h_reg (true h = h_reg + cin)
#pragma unroll
    for (int j = 0; j < 8; ++j) {                    // input layer, C = 0
      v16h b = ldsFrag(b0base + j * 512);
      h[j] = wmma16(a0.v, b, zc);
    }

#pragma unroll
    for (int blk = 0; blk < 2; ++blk) {              // gated residual blocks
      stage_act(actW, h, cin, n, hi, true);          // relu(h_reg + cin)
      v8f t[8];
      gemm128z(t, actRow, sW1 + blk * 16384, n, hi);

      stage_act(actW, t, b1v[blk], n, hi, true);     // relu(t_reg + bl1)
      gemm128z(t, actRow, sW2 + blk * 16384, n, hi); // reuse t

#pragma unroll
      for (int j = 0; j < 8; ++j)                    // h += (t_reg + bl2) * sigmoid(g)
#pragma unroll
        for (int e = 0; e < 8; ++e) h[j][e] += (t[j][e] + b2v[blk][j]) * gsv[blk][j];
    }

    // output layer: 16 cols = one WMMA tile, C = 0, bout folded at publish
    stage_act(actW, h, cin, n, hi, false);
    v8f o;
    {
      v16h a = ldsA(actRow, 0, hi);
      v16h b = ldsB(sWout, n, 0, hi);
      o = wmma16(a, b, zc);
    }
#pragma unroll
    for (int kc = 1; kc < 4; ++kc) {
      v16h a = ldsA(actRow, kc * 32, hi);
      v16h b = ldsB(sWout, n, kc * 32, hi);
      o = wmma16(a, b, o);
    }

    // publish o tile (wave-local region; DS ops are in-order per wave)
#pragma unroll
    for (int r = 0; r < 8; ++r) oW[(r + 8 * hi) * 16 + n] = o[r] + bov;
    lds_fence();

    if (lane < 16) {
      int lr = wave * 16 + lane;
      int row = rowbase + lr;
      if (step < DD) {                               // AR inversion step
#pragma unroll
        for (int d = 0; d < 8; ++d) {
          float mu = oW[lane * 16 + 2 * d + 1];
          float sc = sigmoidf_(oW[lane * 16 + 2 * d] + 2.f) + 0.001f;
          sZ[lr * 8 + d] = (sXin[lr * 8 + d] - mu) / sc;
        }
      } else {                                       // final MADE: ldj only
        float acc = 0.f;
#pragma unroll
        for (int d = 0; d < 8; ++d) {
          float sc = sigmoidf_(oW[lane * 16 + 2 * d] + 2.f) + 0.001f;
          acc -= logf(sc);
        }
#pragma unroll
        for (int d = 0; d < 8; ++d) {
          float zv = sZ[lr * 8 + d];
          z_io[(size_t)row * 8 + d] = zv;
          if (zfin) zfin[(size_t)row * 8 + d] = zv;
        }
        ldj_out[row] = ldj_in[row] + sLogdet + acc;
      }
    }
    lds_fence();
  }
  (void)zero4;
}

// ---------------------------------------------------------------------------
// host launcher
// ---------------------------------------------------------------------------
extern "C" void kernel_launch(void* const* d_in, const int* in_sizes, int n_in,
                              void* d_out, int out_size, void* d_ws, size_t ws_size,
                              hipStream_t stream) {
  (void)in_sizes; (void)n_in; (void)out_size; (void)ws_size;
  const float* x        = (const float*)d_in[0];
  const float* W0       = (const float*)d_in[1];
  const float* b0       = (const float*)d_in[2];
  const float* Wc0      = (const float*)d_in[3];
  const float* bc0      = (const float*)d_in[4];
  const float* Wl1      = (const float*)d_in[5];
  const float* bl1      = (const float*)d_in[6];
  const float* Wl2      = (const float*)d_in[7];
  const float* bl2      = (const float*)d_in[8];
  const float* Wcb      = (const float*)d_in[9];
  const float* bcb      = (const float*)d_in[10];
  const float* Wout     = (const float*)d_in[11];
  const float* bout     = (const float*)d_in[12];
  const float* lu_lower = (const float*)d_in[13];
  const float* lu_upper = (const float*)d_in[14];
  const float* lu_udiag = (const float*)d_in[15];
  const float* lu_bias  = (const float*)d_in[16];
  const int*   perms    = (const int*)d_in[17];

  char* ws = (char*)d_ws;
  float*     z0    = (float*)(ws + 0);                       // NROW*8 f32
  float*     ldjA  = (float*)(ws + 1048576);                 // NROW f32
  float*     ldjB  = (float*)(ws + 1179648);                 // NROW f32
  float*     c0    = (float*)(ws + 1310720);                 // NF*NPOS*128 f32
  float*     gsig  = (float*)(ws + 3407872);                 // NF*2*NPOS*128 f32
  float*     lup   = (float*)(ws + 7602176);                 // NF*80 f32
  _Float16*  wpack = (_Float16*)(ws + 7603456);              // NF*FLOWSZ_H halves

  float* out_z   = (float*)d_out;
  float* out_ldj = out_z + (size_t)NROW * 8;

  patch_kernel<<<1024, 256, 0, stream>>>(x, z0, ldjA);
  prep_kernel<<<NF, 256, 0, stream>>>(W0, Wl1, Wl2, Wout, lu_lower, lu_upper,
                                      lu_udiag, lu_bias, wpack, lup);
  ctx_kernel<<<dim3(NPOS, NF), 128, 0, stream>>>(b0, Wc0, bc0, Wcb, bcb, c0, gsig);

  const float* lin[NF];
  float* lout[NF];
  lin[0] = ldjA; lout[0] = ldjB;     // f=3
  lin[1] = ldjB; lout[1] = ldjA;     // f=2
  lin[2] = ldjA; lout[2] = ldjB;     // f=1
  lin[3] = ldjB; lout[3] = out_ldj;  // f=0

  int li = 0;
  for (int f = NF - 1; f >= 0; --f, ++li) {
    float* zfin = (f == 0) ? out_z : nullptr;
    flow_kernel<<<NROW / 64, 128, LDS_TOTAL, stream>>>(
        z0, lin[li], lout[li], zfin,
        wpack + (size_t)f * FLOWSZ_H,
        c0 + (size_t)f * NPOS * 128,
        gsig + (size_t)f * 2 * NPOS * 128,
        lup + f * 80,
        bl1 + f * 256, bl2 + f * 256, bout + f * 16, perms + f * 8);
  }
}
